// MyModel_75076028334799
// MI455X (gfx1250) — matrix operand order repaired and checked by reading, not compiled, not run on hardware
//
#include <hip/hip_runtime.h>
#include <hip/hip_bf16.h>
#include <math.h>

// Problem constants (fixed by the reference)
#define T_STEPS 60
#define NN      10000
#define EE      160000
#define INC     64
#define HIDC    128
#define OUTC    6

typedef __attribute__((ext_vector_type(16))) _Float16 v16h;
typedef __attribute__((ext_vector_type(8)))  float    v8f;

// Monotone float<->uint encoding so float-max can use integer atomicMax.
__device__ __forceinline__ unsigned fenc(float f) {
  unsigned u = __float_as_uint(f);
  return (u >> 31) ? ~u : (u | 0x80000000u);
}
__device__ __forceinline__ float fdec(unsigned e) {
  return (e >> 31) ? __uint_as_float(e & 0x7fffffffu) : __uint_as_float(~e);
}

// ---------------------------------------------------------------------------
// One-time init: build f16 concatenated weights, zero the max accumulators.
// Wcat1[o][k] : k<64 -> W1_l[o][k], else W1_r[o][k-64]        (128 x 128)
// Wcat2[o][k] : o<6: k<128 -> W2_l[o][k], else W2_r[o][k-128] (16 x 256, pad 0)
// ---------------------------------------------------------------------------
__global__ void k_init(const float* __restrict__ W1l, const float* __restrict__ W1r,
                       const float* __restrict__ W2l, const float* __restrict__ W2r,
                       _Float16* __restrict__ Wc1, _Float16* __restrict__ Wc2,
                       unsigned* __restrict__ zenc) {
  int i = blockIdx.x * blockDim.x + threadIdx.x;
  if (i < 128 * 128) {
    int o = i >> 7, k = i & 127;
    Wc1[i] = (_Float16)((k < 64) ? W1l[o * 64 + k] : W1r[o * 64 + (k - 64)]);
  } else if (i < 128 * 128 + 16 * 256) {
    int j = i - 128 * 128;
    int o = j >> 8, k = j & 255;
    float v = 0.0f;
    if (o < OUTC) v = (k < 128) ? W2l[o * 128 + k] : W2r[o * 128 + (k - 128)];
    Wc2[j] = (_Float16)v;
  } else if (i < 128 * 128 + 16 * 256 + T_STEPS * 8) {
    zenc[i - (128 * 128 + 16 * 256)] = 0u;   // encoding floor (below all reals)
  }
}

__global__ void k_zero(float* __restrict__ p, int n) {
  int i = blockIdx.x * blockDim.x + threadIdx.x;
  if (i < n) p[i] = 0.0f;
}

// ---------------------------------------------------------------------------
// Edge scatter, layer 1: msg1[dst] += x[src] (64ch), deg[dst] += 1
// One thread per (edge, 4-channel chunk).
// ---------------------------------------------------------------------------
__global__ void k_scatter1(const float* __restrict__ x_t,
                           const int* __restrict__ src, const int* __restrict__ dst,
                           float* __restrict__ msg1, float* __restrict__ deg) {
  int tid = blockIdx.x * blockDim.x + threadIdx.x;
  int e = tid >> 4;
  if (e >= EE) return;
  int ch = (tid & 15) << 2;
  int s = src[e], d = dst[e];
  float4 xv = *(const float4*)(x_t + (size_t)s * INC + ch);
  float* mp = msg1 + (size_t)d * INC + ch;
  atomicAdd(mp + 0, xv.x);
  atomicAdd(mp + 1, xv.y);
  atomicAdd(mp + 2, xv.z);
  atomicAdd(mp + 3, xv.w);
  if (ch == 0) atomicAdd(deg + d, 1.0f);
}

// Edge scatter, layer 2: msg2[dst] += h32[src] (128ch)
__global__ void k_scatter2(const float* __restrict__ h32,
                           const int* __restrict__ src, const int* __restrict__ dst,
                           float* __restrict__ msg2) {
  int tid = blockIdx.x * blockDim.x + threadIdx.x;
  int e = tid >> 5;
  if (e >= EE) return;
  int ch = (tid & 31) << 2;
  int s = src[e], d = dst[e];
  float4 hv = *(const float4*)(h32 + (size_t)s * HIDC + ch);
  float* mp = msg2 + (size_t)d * HIDC + ch;
  atomicAdd(mp + 0, hv.x);
  atomicAdd(mp + 1, hv.y);
  atomicAdd(mp + 2, hv.z);
  atomicAdd(mp + 3, hv.w);
}

// ---------------------------------------------------------------------------
// SAGE layer 1 via WMMA: h = relu([mean | x] @ Wcat1^T + b1)
// Block = 256 threads = 8 waves; block owns a 16-row tile, each wave one of
// the 8 16-col output tiles. K = 128 -> 4 x v_wmma_f32_16x16x32_f16.
// ---------------------------------------------------------------------------
__global__ void __launch_bounds__(256)
k_sage1(const float* __restrict__ msg1, const float* __restrict__ deg,
        const float* __restrict__ x_t, const _Float16* __restrict__ Wc1,
        const float* __restrict__ b1,
        float* __restrict__ h32, _Float16* __restrict__ h16) {
  int lane = threadIdx.x & 31;
  int wv   = threadIdx.x >> 5;      // 0..7 -> output col tile
  int r    = lane & 15;
  int hi   = lane >> 4;             // lane half selects K sub-pattern
  int node = blockIdx.x * 16 + r;

  const float* mrow = msg1 + (size_t)node * INC;
  const float* xrow = x_t  + (size_t)node * INC;
  float rd = 1.0f / fmaxf(deg[node], 1.0f);

  v8f acc = {};
#pragma unroll
  for (int kk = 0; kk < 4; ++kk) {
    int kb = kk * 32;
    v16h a;
    if (kb < 64) {                  // K 0..63 : mean = msg1 * rdeg
      const float* p0 = mrow + kb + hi * 8;
      const float* p1 = mrow + kb + 16 + hi * 8;
#pragma unroll
      for (int j = 0; j < 8; ++j) {
        a[j]     = (_Float16)(p0[j] * rd);
        a[8 + j] = (_Float16)(p1[j] * rd);
      }
    } else {                        // K 64..127 : raw x
      const float* p0 = xrow + (kb - 64) + hi * 8;
      const float* p1 = xrow + (kb - 64) + 16 + hi * 8;
#pragma unroll
      for (int j = 0; j < 8; ++j) {
        a[j]     = (_Float16)p0[j];
        a[8 + j] = (_Float16)p1[j];
      }
    }
    // B: lane holds K {hi*8..hi*8+7, 16+hi*8..16+hi*8+7} of Wcat1 row (out-channel) wv*16+r
    const _Float16* brow = Wc1 + (size_t)(wv * 16 + r) * HIDC + kb + hi * 8;
    v16h b;
#pragma unroll
    for (int j = 0; j < 16; ++j) b[j] = brow[j + (j & 8)];
    acc = __builtin_amdgcn_wmma_f32_16x16x32_f16(false, a, false, b,
                                                 (short)0, acc, false, false);
  }

  int col = wv * 16 + r;
  float bias = b1[col];
#pragma unroll
  for (int v = 0; v < 8; ++v) {
    int m = blockIdx.x * 16 + v + hi * 8;     // C/D layout: rows v / v+8
    float val = fmaxf(acc[v] + bias, 0.0f);   // ReLU
    h32[(size_t)m * HIDC + col] = val;
    h16[(size_t)m * HIDC + col] = (_Float16)val;
  }
}

// ---------------------------------------------------------------------------
// SAGE layer 2 + global max pool: out = [mean2 | h] @ Wcat2^T + b2 (6 cols),
// K = 256 -> 8 WMMAs; fused column max via encoded atomicMax into zenc[t].
// Block = 128 threads = 4 waves, one 16-row tile per wave.
// ---------------------------------------------------------------------------
__global__ void __launch_bounds__(128)
k_sage2max(const float* __restrict__ msg2, const float* __restrict__ deg,
           const _Float16* __restrict__ h16, const _Float16* __restrict__ Wc2,
           const float* __restrict__ b2, unsigned* __restrict__ zenc, int t) {
  int lane  = threadIdx.x & 31;
  int wv    = threadIdx.x >> 5;
  int mtile = blockIdx.x * 4 + wv;
  if (mtile >= (NN / 16)) return;
  int r  = lane & 15;
  int hi = lane >> 4;
  int node = mtile * 16 + r;

  const float*    mrow = msg2 + (size_t)node * HIDC;
  const _Float16* hrow = h16  + (size_t)node * HIDC;
  float rd = 1.0f / fmaxf(deg[node], 1.0f);

  v8f acc = {};
#pragma unroll
  for (int kk = 0; kk < 8; ++kk) {
    int kb = kk * 32;
    v16h a;
    if (kb < 128) {                 // K 0..127 : mean2
      const float* p0 = mrow + kb + hi * 8;
      const float* p1 = mrow + kb + 16 + hi * 8;
#pragma unroll
      for (int j = 0; j < 8; ++j) {
        a[j]     = (_Float16)(p0[j] * rd);
        a[8 + j] = (_Float16)(p1[j] * rd);
      }
    } else {                        // K 128..255 : h (already f16)
      const _Float16* p0 = hrow + (kb - 128) + hi * 8;
      const _Float16* p1 = hrow + (kb - 128) + 16 + hi * 8;
#pragma unroll
      for (int j = 0; j < 8; ++j) {
        a[j]     = p0[j];
        a[8 + j] = p1[j];
      }
    }
    const _Float16* brow = Wc2 + (size_t)r * 256 + kb + hi * 8;
    v16h b;
#pragma unroll
    for (int j = 0; j < 16; ++j) b[j] = brow[j + (j & 8)];
    acc = __builtin_amdgcn_wmma_f32_16x16x32_f16(false, a, false, b,
                                                 (short)0, acc, false, false);
  }

  float bias = (r < OUTC) ? b2[r] : 0.0f;
  float vmax = -3.4e38f;
#pragma unroll
  for (int v = 0; v < 8; ++v) vmax = fmaxf(vmax, acc[v] + bias);
  float other = __shfl_xor(vmax, 16, 32);     // combine the two row-halves
  vmax = fmaxf(vmax, other);
  if (hi == 0 && r < OUTC)
    atomicMax(zenc + t * 8 + r, fenc(vmax));
}

// ---------------------------------------------------------------------------
// Head: decode z, GRU (h0 = 0 so whh terms vanish except r*bhn), NCE + acc,
// hidden = gru_out[:30]. Single block.
// ---------------------------------------------------------------------------
__global__ void __launch_bounds__(256)
k_head(const unsigned* __restrict__ zenc, const float* __restrict__ wih,
       const float* __restrict__ bih, const float* __restrict__ bhh,
       float* __restrict__ out) {
  __shared__ float zl[T_STEPS][OUTC];
  __shared__ float go[T_STEPS][OUTC];
  __shared__ float s_nce;
  __shared__ int   s_cor;
  int tid = threadIdx.x;
  if (tid == 0) { s_nce = 0.0f; s_cor = 0; }
  for (int i = tid; i < T_STEPS * OUTC; i += blockDim.x) {
    int tt = i / OUTC, n = i % OUTC;
    zl[tt][n] = fdec(zenc[tt * 8 + n]);
  }
  __syncthreads();

  for (int i = tid; i < T_STEPS * OUTC; i += blockDim.x) {
    int tt = i / OUTC, n = i % OUTC;
    float xr = 0.f, xz = 0.f, xn = 0.f;
#pragma unroll
    for (int k = 0; k < OUTC; ++k) {
      float zk = zl[tt][k];
      xr += zk * wih[(0 * OUTC + n) * OUTC + k];
      xz += zk * wih[(1 * OUTC + n) * OUTC + k];
      xn += zk * wih[(2 * OUTC + n) * OUTC + k];
    }
    float rg = 1.0f / (1.0f + expf(-(xr + bih[n] + bhh[n])));
    float ug = 1.0f / (1.0f + expf(-(xz + bih[OUTC + n] + bhh[OUTC + n])));
    float ng = tanhf(xn + bih[2 * OUTC + n] + rg * bhh[2 * OUTC + n]);
    go[tt][n] = (1.0f - ug) * ng;
  }
  __syncthreads();

  // t0=10, t1=30, cnt=20; i in 1..12; j_off = {0, 11..17}
  if (tid < 20 * 12) {
    int ti = tid / 12, i = 1 + tid % 12;
    int tt = 10 + ti;
    float l[8];
#pragma unroll
    for (int s = 0; s < 8; ++s) {
      int j = (s == 0) ? 0 : (10 + s);
      int idx = tt + i + j;
      float d = 0.f;
#pragma unroll
      for (int k = 0; k < OUTC; ++k) d += zl[idx][k] * go[tt][k];
      l[s] = d;
    }
    float ml = l[0];
#pragma unroll
    for (int s = 1; s < 8; ++s) ml = fmaxf(ml, l[s]);
    float se = 0.f;
#pragma unroll
    for (int s = 0; s < 8; ++s) se += expf(l[s] - ml);
    atomicAdd(&s_nce, l[0] - ml - logf(se));
    bool ok = true;
#pragma unroll
    for (int s = 1; s < 8; ++s) ok = ok && (l[0] >= l[s]);
    if (ok) atomicAdd(&s_cor, 1);
  }
  __syncthreads();

  if (tid == 0) {
    out[0] = s_nce / (-240.0f);          // nce = sum / (-cnt*TIMESTAMP)
    out[1] = (float)s_cor / 240.0f;      // acc
  }
  for (int i = tid; i < 30 * OUTC; i += blockDim.x)
    out[2 + i] = go[i / OUTC][i % OUTC]; // hidden [1,30,6]
}

// ---------------------------------------------------------------------------
extern "C" void kernel_launch(void* const* d_in, const int* in_sizes, int n_in,
                              void* d_out, int out_size, void* d_ws, size_t ws_size,
                              hipStream_t stream) {
  (void)in_sizes; (void)n_in; (void)out_size; (void)ws_size;
  const float* x_seq = (const float*)d_in[0];
  const int*   ei    = (const int*)d_in[1];
  const float* W1l   = (const float*)d_in[2];
  const float* W1r   = (const float*)d_in[3];
  const float* b1    = (const float*)d_in[4];
  const float* W2l   = (const float*)d_in[5];
  const float* W2r   = (const float*)d_in[6];
  const float* b2    = (const float*)d_in[7];
  const float* gwih  = (const float*)d_in[8];
  const float* gbih  = (const float*)d_in[10];
  const float* gbhh  = (const float*)d_in[11];

  // Workspace layout (~15.5 MB total). deg|msg1|msg2 contiguous for one-shot zero.
  char* ws = (char*)d_ws;
  float* deg  = (float*)ws;                         // NN floats
  float* msg1 = deg + NN;                           // NN*64
  float* msg2 = msg1 + (size_t)NN * INC;            // NN*128
  const size_t zero_count = (size_t)NN * (1 + INC + HIDC);   // 1,930,000
  size_t off = ((zero_count * 4 + 255) / 256) * 256;
  float* h32 = (float*)(ws + off);        off += (size_t)NN * HIDC * 4;
  _Float16* h16 = (_Float16*)(ws + off);  off += (size_t)NN * HIDC * 2;
  _Float16* Wc1 = (_Float16*)(ws + off);  off += 128 * 128 * 2;
  _Float16* Wc2 = (_Float16*)(ws + off);  off += 16 * 256 * 2;
  unsigned* zenc = (unsigned*)(ws + off); off += T_STEPS * 8 * 4;

  k_init<<<(128 * 128 + 16 * 256 + T_STEPS * 8 + 255) / 256, 256, 0, stream>>>(
      W1l, W1r, W2l, W2r, Wc1, Wc2, zenc);

  for (int t = 0; t < T_STEPS; ++t) {
    const float* x_t = x_seq + (size_t)t * NN * INC;
    const int*   src = ei + (size_t)t * 2 * EE;
    const int*   dst = src + EE;
    k_zero<<<(int)((zero_count + 255) / 256), 256, 0, stream>>>(deg, (int)zero_count);
    k_scatter1<<<(EE * 16) / 256, 256, 0, stream>>>(x_t, src, dst, msg1, deg);
    k_sage1<<<NN / 16, 256, 0, stream>>>(msg1, deg, x_t, Wc1, b1, h32, h16);
    k_scatter2<<<(EE * 32) / 256, 256, 0, stream>>>(h32, src, dst, msg2);
    k_sage2max<<<(NN / 16 + 3) / 4, 128, 0, stream>>>(msg2, deg, h16, Wc2, b2, zenc, t);
  }

  k_head<<<1, 256, 0, stream>>>(zenc, gwih, gbih, gbhh, (float*)d_out);
}
